// VariableSelectionNetwork_82257213653636
// MI455X (gfx1250) — matrix operand order repaired and checked by reading, not compiled, not run on hardware
//
#include <hip/hip_runtime.h>
#include <hip/hip_bf16.h>
#include <math.h>

// ---------------- problem constants ----------------
constexpr int BATCH = 4096;     // batch rows
constexpr int NVAR  = 16;       // variables
constexpr int D     = 256;      // per-var dim (DIN=DH=DOUT)
constexpr int NF    = NVAR * D; // 4096 flattened features

// ---------------- vector types ----------------
typedef __attribute__((ext_vector_type(4)))  __bf16 bf16x4;
typedef __attribute__((ext_vector_type(8)))  __bf16 bf16x8;
typedef __attribute__((ext_vector_type(16))) __bf16 bf16x16;
typedef __attribute__((ext_vector_type(8)))  float  f32x8;
typedef __attribute__((ext_vector_type(4)))  int    i32x4;

// ---------------- CDNA5 async global->LDS path (ASYNCcnt-tracked) ----------------
#if defined(__AMDGCN__) && __has_builtin(__builtin_amdgcn_global_load_async_to_lds_b128)
#define HAVE_ASYNC_LDS 1
#else
#define HAVE_ASYNC_LDS 0
#endif

#if HAVE_ASYNC_LDS
typedef __attribute__((address_space(1))) i32x4 gas_i32x4;   // global
typedef __attribute__((address_space(3))) i32x4 las_i32x4;   // LDS

__device__ __forceinline__ void async_copy16(const void* g, void* l) {
    // global_load_async_to_lds_b128: 16B per lane, LDS written directly (no VGPR round trip)
    __builtin_amdgcn_global_load_async_to_lds_b128(
        (gas_i32x4*)(unsigned long long)g,
        (las_i32x4*)(unsigned int)(unsigned long long)l,
        0, 0);
}
__device__ __forceinline__ void wait_async_le8() {
#if __has_builtin(__builtin_amdgcn_s_wait_asynccnt)
    __builtin_amdgcn_s_wait_asynccnt(8);
#else
    asm volatile("s_wait_asynccnt 0x8" ::: "memory");
#endif
}
__device__ __forceinline__ void wait_async_0() {
#if __has_builtin(__builtin_amdgcn_s_wait_asynccnt)
    __builtin_amdgcn_s_wait_asynccnt(0);
#else
    asm volatile("s_wait_asynccnt 0x0" ::: "memory");
#endif
}
#endif

// ---------------- f32 -> bf16 convert ----------------
__global__ __launch_bounds__(256) void f32_to_bf16_kernel(const float* __restrict__ in,
                                                          __bf16* __restrict__ out, long n) {
    long i = ((long)blockIdx.x * blockDim.x + threadIdx.x) * 4;
    if (i + 3 < n) {
        float4 v = *(const float4*)(in + i);
        bf16x4 o;
        o[0] = (__bf16)v.x; o[1] = (__bf16)v.y; o[2] = (__bf16)v.z; o[3] = (__bf16)v.w;
        *(bf16x4*)(out + i) = o;
    }
}

// ---------------- f32 [K,M] -> bf16 [M,K] transpose (batched) ----------------
__global__ __launch_bounds__(256) void transpose_to_bf16_kernel(const float* __restrict__ in,
                                                                __bf16* __restrict__ out,
                                                                int K, int M,
                                                                long inStride, long outStride) {
    __shared__ float tile[32][33];
    in  += (long)blockIdx.z * inStride;
    out += (long)blockIdx.z * outStride;
    const int k0 = blockIdx.y * 32, m0 = blockIdx.x * 32;
    const int tx = threadIdx.x, ty = threadIdx.y;   // 32 x 8
#pragma unroll
    for (int i = 0; i < 32; i += 8)
        tile[ty + i][tx] = in[(long)(k0 + ty + i) * M + m0 + tx];
    __syncthreads();
#pragma unroll
    for (int i = 0; i < 32; i += 8)
        out[(long)(m0 + ty + i) * K + k0 + tx] = (__bf16)tile[tx][ty + i];
}

// ---------------- WMMA GEMM: C = A[M,K](bf16) * Wt[N,K](bf16)^T + bias ----------------
// MODE 0: ELU -> bf16 out; MODE 1: identity -> bf16 out; MODE 2: identity -> f32 out
constexpr int BM = 128, BN = 128, BK = 64, KPAD = 72; // 72*2B = 144B, multiple of 16 -> aligned b128
constexpr int TILE_ELEMS = 2 * BM * KPAD;             // A tile | W tile per buffer

template <int MODE>
__global__ __launch_bounds__(256) void gemm_kernel(const __bf16* __restrict__ A, long aBatch, int lda,
                                                   const __bf16* __restrict__ Wt, long wBatch,
                                                   const float* __restrict__ bias, long bBatch,
                                                   void* __restrict__ Cv, long cBatch, int ldc,
                                                   int K) {
#if HAVE_ASYNC_LDS
    __shared__ __bf16 lds[3][TILE_ELEMS];   // triple buffer, 108 KB (one barrier per K-tile)
#else
    __shared__ __bf16 lds[2][TILE_ELEMS];   // double buffer, 72 KB
#endif
    const int tid = threadIdx.x;
    const int z = blockIdx.z;
    A    += (long)z * aBatch;
    Wt   += (long)z * wBatch;
    bias += (long)z * bBatch;
    const int rowBase = blockIdx.y * BM;
    const int colBase = blockIdx.x * BN;

    const int lane = tid & 31;
    const int wid  = tid >> 5;
    const int wRow = (wid >> 1) * 32;  // 0..96
    const int wCol = (wid & 1) * 64;   // 0,64
    const int l15  = lane & 15;
    const int lh   = lane >> 4;

    f32x8 acc[2][4];
#pragma unroll
    for (int i = 0; i < 2; ++i)
#pragma unroll
        for (int j = 0; j < 4; ++j) {
            f32x8 zf = {0.f, 0.f, 0.f, 0.f, 0.f, 0.f, 0.f, 0.f};
            acc[i][j] = zf;
        }

    const int nTiles = K / BK;

    // ------- WMMA compute on one resident LDS tile -------
    auto compute = [&](const __bf16* la, const __bf16* lw) {
#pragma unroll
        for (int ks = 0; ks < 2; ++ks) {
            const int k0 = ks * 32;
            // A fragment (ISA 16-bit A 16x32): lanes 0-15 K 0-7 & 16-23; lanes 16-31 K 8-15 & 24-31
            bf16x16 aF[2];
#pragma unroll
            for (int rt = 0; rt < 2; ++rt) {
                const __bf16* p = la + (wRow + rt * 16 + l15) * KPAD + k0 + lh * 8;
                bf16x8 lo = *(const bf16x8*)p;
                bf16x8 hi = *(const bf16x8*)(p + 16);
                aF[rt] = __builtin_shufflevector(lo, hi, 0, 1, 2, 3, 4, 5, 6, 7,
                                                 8, 9, 10, 11, 12, 13, 14, 15);
            }
            // B fragment (dense 32x16): lanes 0-15 K 0-7 & 16-23; lanes 16-31 K 8-15 & 24-31
#pragma unroll
            for (int ct = 0; ct < 4; ++ct) {
                const __bf16* p = lw + (wCol + ct * 16 + l15) * KPAD + k0 + lh * 8;
                bf16x8 lo = *(const bf16x8*)p;
                bf16x8 hi = *(const bf16x8*)(p + 16);
                bf16x16 bF = __builtin_shufflevector(lo, hi, 0, 1, 2, 3, 4, 5, 6, 7,
                                                     8, 9, 10, 11, 12, 13, 14, 15);
                acc[0][ct] = __builtin_amdgcn_wmma_f32_16x16x32_bf16(
                    false, aF[0], false, bF, (short)0, acc[0][ct], false, false);
                acc[1][ct] = __builtin_amdgcn_wmma_f32_16x16x32_bf16(
                    false, aF[1], false, bF, (short)0, acc[1][ct], false, false);
            }
        }
    };

#if HAVE_ASYNC_LDS
    // ------- async global->LDS staging: 8 b128 async ops per wave per tile -------
    auto stage = [&](int t) {
        const int buf = t % 3;
        const int kt = t * BK;
        __bf16* la = &lds[buf][0];
        __bf16* lw = &lds[buf][BM * KPAD];
#pragma unroll
        for (int i = 0; i < 4; ++i) {
            int c = i * 256 + tid;            // 1024 16B chunks per tile
            int r = c >> 3;                   // 0..127
            int kc = (c & 7) * 8;             // 0..56
            async_copy16(A + (long)(rowBase + r) * lda + kt + kc, la + r * KPAD + kc);
            async_copy16(Wt + (long)(colBase + r) * K + kt + kc, lw + r * KPAD + kc);
        }
    };

    stage(0);
    if (nTiles > 1) stage(1);
    for (int t = 0; t < nTiles; ++t) {
        // async loads complete in order: keeping <=8 outstanding guarantees tile t is in LDS
        if (t + 1 < nTiles) wait_async_le8();
        else                wait_async_0();
        __syncthreads();                       // publish tile t to all waves
        if (t + 2 < nTiles) stage(t + 2);      // buf (t+2)%3: no reader after the barrier
        const int cur = t % 3;
        compute(&lds[cur][0], &lds[cur][BM * KPAD]);
    }
#else
    // ------- fallback: global->VGPR->LDS double-buffered staging -------
    bf16x8 regA[4], regW[4];
    auto loadRegs = [&](int kt) {
#pragma unroll
        for (int i = 0; i < 4; ++i) {
            int c = i * 256 + tid;
            int r = c >> 3;
            int kc = (c & 7) * 8;
            regA[i] = *(const bf16x8*)(A + (long)(rowBase + r) * lda + kt + kc);
            regW[i] = *(const bf16x8*)(Wt + (long)(colBase + r) * K + kt + kc);
        }
    };
    auto storeRegs = [&](int buf) {
        __bf16* la = &lds[buf][0];
        __bf16* lw = &lds[buf][BM * KPAD];
#pragma unroll
        for (int i = 0; i < 4; ++i) {
            int c = i * 256 + tid;
            int r = c >> 3;
            int kc = (c & 7) * 8;
            *(bf16x8*)(la + r * KPAD + kc) = regA[i];
            *(bf16x8*)(lw + r * KPAD + kc) = regW[i];
        }
    };

    loadRegs(0);
    storeRegs(0);
    __syncthreads();
    for (int t = 0; t < nTiles; ++t) {
        const int cur = t & 1;
        if (t + 1 < nTiles) loadRegs((t + 1) * BK);
        compute(&lds[cur][0], &lds[cur][BM * KPAD]);
        if (t + 1 < nTiles) storeRegs((t + 1) & 1);
        __syncthreads();
    }
#endif

    // Epilogue. D layout: lanes 0-15 -> M=j, N=lane; lanes 16-31 -> M=8+j, N=lane-16
    const long cz = (long)z * cBatch;
#pragma unroll
    for (int ct = 0; ct < 4; ++ct) {
        const int gcol = colBase + wCol + ct * 16 + l15;
        const float bv = bias[gcol];
#pragma unroll
        for (int rt = 0; rt < 2; ++rt) {
#pragma unroll
            for (int j = 0; j < 8; ++j) {
                const int grow = rowBase + wRow + rt * 16 + lh * 8 + j;
                float v = acc[rt][ct][j] + bv;
                if (MODE == 0) v = v > 0.f ? v : (expf(v) - 1.f);   // ELU
                const long idx = (long)grow * ldc + cz + gcol;
                if (MODE == 2) ((float*)Cv)[idx] = v;
                else           ((__bf16*)Cv)[idx] = (__bf16)v;
            }
        }
    }
}

// ---------------- block reduction (256 threads, wave32) ----------------
__device__ inline void blockReduce2(float& a, float& b, float* rs, float* rq, int tid) {
#pragma unroll
    for (int o = 16; o > 0; o >>= 1) {
        a += __shfl_xor(a, o, 32);
        b += __shfl_xor(b, o, 32);
    }
    if ((tid & 31) == 0) { rs[tid >> 5] = a; rq[tid >> 5] = b; }
    __syncthreads();
    float ta = 0.f, tb = 0.f;
#pragma unroll
    for (int i = 0; i < 8; ++i) { ta += rs[i]; tb += rq[i]; }
    a = ta; b = tb;
    __syncthreads();
}

// ---------------- GLU + residual + LN(4096) + softmax over N ----------------
__global__ __launch_bounds__(256) void ln_softmax_kernel(const float* __restrict__ x,
                                                         const float* __restrict__ e4,
                                                         const float* __restrict__ e5,
                                                         const float* __restrict__ lng,
                                                         const float* __restrict__ lnb,
                                                         float* __restrict__ wsel) {
    __shared__ float cs[NF];
    __shared__ float rs[8], rq[8];
    const int t = threadIdx.x;
    const long base = (long)blockIdx.x * NF;
    float s[16];
    float sum = 0.f, sq = 0.f;
#pragma unroll
    for (int j = 0; j < 16; ++j) {
        const int i = j * 256 + t;
        const float g = e5[base + i] / (1.f + expf(-e4[base + i]));   // sigmoid(e4)*e5
        const float v = x[base + i] + g;
        s[j] = v; sum += v; sq += v * v;
    }
    blockReduce2(sum, sq, rs, rq, t);
    const float mean = sum * (1.f / NF);
    const float var  = sq * (1.f / NF) - mean * mean;
    const float rstd = rsqrtf(var + 1e-5f);
#pragma unroll
    for (int j = 0; j < 16; ++j) {
        const int i = j * 256 + t;
        cs[i] = (s[j] - mean) * rstd * lng[i] + lnb[i];
    }
    __syncthreads();
    // softmax over variable axis (16 strided values per feature d = t)
    float vv[NVAR];
    float m = -3.4e38f;
#pragma unroll
    for (int n = 0; n < NVAR; ++n) { vv[n] = cs[n * 256 + t]; m = fmaxf(m, vv[n]); }
    float se = 0.f;
#pragma unroll
    for (int n = 0; n < NVAR; ++n) { vv[n] = expf(vv[n] - m); se += vv[n]; }
    const float inv = 1.f / se;
#pragma unroll
    for (int n = 0; n < NVAR; ++n) wsel[base + n * 256 + t] = vv[n] * inv;
}

// ---------------- per-var GLU + residual + LN(256) + weighted combine ----------------
__global__ __launch_bounds__(256) void ln_combine_kernel(const float* __restrict__ x,
                                                         const float* __restrict__ e4,
                                                         const float* __restrict__ e5,
                                                         const float* __restrict__ lng,
                                                         const float* __restrict__ lnb,
                                                         const float* __restrict__ wsel,
                                                         float* __restrict__ out) {
    __shared__ float rs[8], rq[8];
    const int t = threadIdx.x;
    const long base = (long)blockIdx.x * NF;
    float accum = 0.f;
    for (int n = 0; n < NVAR; ++n) {
        const int i = n * 256 + t;
        const float g = e5[base + i] / (1.f + expf(-e4[base + i]));
        const float v = x[base + i] + g;
        float sum = v, sq = v * v;
        blockReduce2(sum, sq, rs, rq, t);
        const float mean = sum * (1.f / 256.f);
        const float var  = sq * (1.f / 256.f) - mean * mean;
        const float rstd = rsqrtf(var + 1e-5f);
        const float yv = (v - mean) * rstd * lng[i] + lnb[i];
        accum += wsel[base + i] * yv;
    }
    out[(long)blockIdx.x * 256 + t] = accum;
}

// ---------------- launcher ----------------
extern "C" void kernel_launch(void* const* d_in, const int* in_sizes, int n_in,
                              void* d_out, int out_size, void* d_ws, size_t ws_size,
                              hipStream_t stream) {
    (void)in_sizes; (void)n_in; (void)out_size; (void)ws_size;
    // inputs in setup_inputs() order
    const float* x    = (const float*)d_in[0];
    const float* vW2  = (const float*)d_in[1];
    const float* vb2  = (const float*)d_in[2];
    const float* vW1  = (const float*)d_in[3];
    const float* vb1  = (const float*)d_in[4];
    const float* vW4  = (const float*)d_in[5];
    const float* vb4  = (const float*)d_in[6];
    const float* vW5  = (const float*)d_in[7];
    const float* vb5  = (const float*)d_in[8];
    const float* vlng = (const float*)d_in[9];
    const float* vlnb = (const float*)d_in[10];
    const float* cW2  = (const float*)d_in[11];
    const float* cb2  = (const float*)d_in[12];
    const float* cW1  = (const float*)d_in[13];
    const float* cb1  = (const float*)d_in[14];
    const float* cW4  = (const float*)d_in[15];
    const float* cb4  = (const float*)d_in[16];
    const float* cW5  = (const float*)d_in[17];
    const float* cb5  = (const float*)d_in[18];
    const float* clng = (const float*)d_in[19];
    const float* clnb = (const float*)d_in[20];
    float* out = (float*)d_out;

    // ---- workspace carve-up (256B aligned) ----
    char* ws = (char*)d_ws;
    size_t off = 0;
    auto carve = [&](size_t bytes) -> void* {
        void* p = ws + off;
        off += (bytes + 255) & ~(size_t)255;
        return p;
    };
    const size_t big_bf16 = (size_t)NF * NF * 2;        // 4096x4096 bf16
    const size_t big_f32  = (size_t)NF * NF * 4;        // 4096x4096 f32
    const size_t var_bf16 = (size_t)NVAR * D * D * 2;   // 16x256x256 bf16

    __bf16* xb   = (__bf16*)carve(big_bf16);
    __bf16* cW2t = (__bf16*)carve(big_bf16);
    __bf16* cW1t = (__bf16*)carve(big_bf16);
    __bf16* cW4t = (__bf16*)carve(big_bf16);
    __bf16* cW5t = (__bf16*)carve(big_bf16);
    __bf16* vW2t = (__bf16*)carve(var_bf16);
    __bf16* vW1t = (__bf16*)carve(var_bf16);
    __bf16* vW4t = (__bf16*)carve(var_bf16);
    __bf16* vW5t = (__bf16*)carve(var_bf16);
    __bf16* hb   = (__bf16*)carve(big_bf16);            // h  (reused for hv)
    __bf16* e1b  = (__bf16*)carve(big_bf16);            // e1 (reused for e1v)
    float*  e4f  = (float*)carve(big_f32);              // e4 (reused for e4v)
    float*  e5f  = (float*)carve(big_f32);              // e5 (reused for e5v)
    float*  wsel = (float*)carve(big_f32);              // softmax weights

    const dim3 tb256(256);

    // 1) x -> bf16 (flat layout == [B, N*DIN])
    {
        long n = (long)BATCH * NF;
        f32_to_bf16_kernel<<<dim3((unsigned)(n / (256 * 4))), tb256, 0, stream>>>(x, xb, n);
    }
    // 2) transpose concat weights [4096,4096] f32 -> [4096,4096] bf16 (Wt[m][k])
    {
        dim3 g(NF / 32, NF / 32, 1), b(32, 8);
        transpose_to_bf16_kernel<<<g, b, 0, stream>>>(cW2, cW2t, NF, NF, 0, 0);
        transpose_to_bf16_kernel<<<g, b, 0, stream>>>(cW1, cW1t, NF, NF, 0, 0);
        transpose_to_bf16_kernel<<<g, b, 0, stream>>>(cW4, cW4t, NF, NF, 0, 0);
        transpose_to_bf16_kernel<<<g, b, 0, stream>>>(cW5, cW5t, NF, NF, 0, 0);
    }
    // 3) transpose per-var weights [16,256,256]
    {
        dim3 g(D / 32, D / 32, NVAR), b(32, 8);
        long s = (long)D * D;
        transpose_to_bf16_kernel<<<g, b, 0, stream>>>(vW2, vW2t, D, D, s, s);
        transpose_to_bf16_kernel<<<g, b, 0, stream>>>(vW1, vW1t, D, D, s, s);
        transpose_to_bf16_kernel<<<g, b, 0, stream>>>(vW4, vW4t, D, D, s, s);
        transpose_to_bf16_kernel<<<g, b, 0, stream>>>(vW5, vW5t, D, D, s, s);
    }
    // 4) concat GRN GEMM chain (M=4096, N=4096, K=4096)
    {
        dim3 g(NF / BN, BATCH / BM, 1);
        gemm_kernel<0><<<g, tb256, 0, stream>>>(xb, 0, NF, cW2t, 0, cb2, 0, hb, 0, NF, NF);   // h = elu(..)
        gemm_kernel<1><<<g, tb256, 0, stream>>>(hb, 0, NF, cW1t, 0, cb1, 0, e1b, 0, NF, NF);  // e1
        gemm_kernel<2><<<g, tb256, 0, stream>>>(e1b, 0, NF, cW4t, 0, cb4, 0, e4f, 0, NF, NF); // e4 (f32)
        gemm_kernel<2><<<g, tb256, 0, stream>>>(e1b, 0, NF, cW5t, 0, cb5, 0, e5f, 0, NF, NF); // e5 (f32)
    }
    // 5) GLU + residual + LN(4096) + softmax over variables -> wsel
    ln_softmax_kernel<<<dim3(BATCH), tb256, 0, stream>>>(x, e4f, e5f, clng, clnb, wsel);
    // 6) per-variable GRN chain (batched z=0..15, M=4096, N=256, K=256; column-sliced in/out)
    {
        dim3 g(D / BN, BATCH / BM, NVAR);
        long wS = (long)D * D;
        gemm_kernel<0><<<g, tb256, 0, stream>>>(xb, D, NF, vW2t, wS, vb2, D, hb, D, NF, D);   // hv
        gemm_kernel<1><<<g, tb256, 0, stream>>>(hb, D, NF, vW1t, wS, vb1, D, e1b, D, NF, D);  // e1v
        gemm_kernel<2><<<g, tb256, 0, stream>>>(e1b, D, NF, vW4t, wS, vb4, D, e4f, D, NF, D); // e4v
        gemm_kernel<2><<<g, tb256, 0, stream>>>(e1b, D, NF, vW5t, wS, vb5, D, e5f, D, NF, D); // e5v
    }
    // 7) per-var GLU + residual + LN(256) + weighted combine -> out [4096,256]
    ln_combine_kernel<<<dim3(BATCH), tb256, 0, stream>>>(x, e4f, e5f, vlng, vlnb, wsel, out);
}
